// Stage_32066225831987
// MI455X (gfx1250) — matrix operand order repaired and checked by reading, not compiled, not run on hardware
//
#include <hip/hip_runtime.h>

#define BB 4
#define CC 256
#define NN 4096
#define PP 64
#define SS 5

typedef __attribute__((ext_vector_type(16))) __bf16          v16bf;
typedef __attribute__((ext_vector_type(8)))  float           v8f;
typedef __attribute__((ext_vector_type(16))) unsigned short  u16x16;
typedef __attribute__((ext_vector_type(8)))  unsigned short  u16x8;

static __device__ __forceinline__ unsigned short f2bf(float f) {
    unsigned u = __builtin_bit_cast(unsigned, f);
    u += 0x7FFFu + ((u >> 16) & 1u);   // round-to-nearest-even
    return (unsigned short)(u >> 16);
}
static __device__ __forceinline__ float bf2f(unsigned short u) {
    unsigned v = ((unsigned)u) << 16;
    return __builtin_bit_cast(float, v);
}

static __device__ __forceinline__ v8f wmma_bf16(u16x16 a, u16x16 b, v8f c) {
    return __builtin_amdgcn_wmma_f32_16x16x32_bf16(
        false, __builtin_bit_cast(v16bf, a),
        false, __builtin_bit_cast(v16bf, b),
        (short)0, c, false, false);
}

// A fragment: 16x32 bf16 tile of row-major [M,K] matrix at (m0,k0), row stride lda.
// Lane l: row = m0 + (l&15), half h = l>>4; K = k0+16h+e (contiguous).
static __device__ __forceinline__ u16x16 load_a_frag(const unsigned short* A, int lda,
                                                     int m0, int k0) {
    int lane = threadIdx.x & 31;
    int row  = m0 + (lane & 15);
    int h    = lane >> 4;
    const unsigned short* p = A + (size_t)row * lda + k0 + 16 * h;
    u16x8 lo = *(const u16x8*)(p);
    u16x8 hi = *(const u16x8*)(p + 8);
    u16x16 r;
#pragma unroll
    for (int i = 0; i < 8; ++i) { r[i] = lo[i]; r[i + 8] = hi[i]; }
    return r;
}

// B fragment: 32x16 bf16 tile of B at (k0,n0), read from TRANSPOSED storage Bt=[N,K]
// row-major (row stride ldbt). Lane l: col n = n0+(l&15), h = l>>4; K = k0+16h+e.
static __device__ __forceinline__ u16x16 load_bt_frag(const unsigned short* Bt, int ldbt,
                                                      int k0, int n0) {
    int lane = threadIdx.x & 31;
    int col  = n0 + (lane & 15);
    int h    = lane >> 4;
    return *(const u16x16*)(Bt + (size_t)col * ldbt + k0 + 16 * h);
}

// Store 16x16 f32 D tile TRANSPOSED as bf16 into Ct (row = D-col, col = D-row).
// D layout: lane l holds col n=(l&15), rows m = r+8h.
static __device__ __forceinline__ void store_dT_bf16(unsigned short* Ct, int ldct,
                                                     int colOrig, int rowOrig, v8f a) {
    int lane = threadIdx.x & 31;
    int n    = colOrig + (lane & 15);
    int h    = lane >> 4;
    u16x8 o;
#pragma unroll
    for (int r = 0; r < 8; ++r) o[r] = f2bf(a[r]);
    *(u16x8*)(Ct + (size_t)n * ldct + rowOrig + 8 * h) = o;
}

// ---------------- prep: weights -> bf16, BN folded ----------------
__global__ void prep_params(const float* t_w, const float* p_w, const float* g_w,
                            const float* w1, const float* w2, const float* gamma,
                            const float* beta, const float* rm, const float* rv,
                            unsigned short* twb, unsigned short* pwb, unsigned short* gwb,
                            unsigned short* w1b, unsigned short* w2b,
                            float* scale, float* shift) {
    int i = blockIdx.x * blockDim.x + threadIdx.x;
    int st = gridDim.x * blockDim.x;
    for (int k = i; k < PP * CC; k += st) { twb[k] = f2bf(t_w[k]); pwb[k] = f2bf(p_w[k]); }
    for (int k = i; k < SS * PP * CC; k += st) gwb[k] = f2bf(g_w[k]);
    for (int k = i; k < SS * CC * PP; k += st) { w1b[k] = f2bf(w1[k]); w2b[k] = f2bf(w2[k]); }
    for (int k = i; k < SS * CC; k += st) {
        float sc = gamma[k] * rsqrtf(rv[k] + 1e-5f);
        scale[k] = sc;
        shift[k] = beta[k] - rm[k] * sc;
    }
}

// ---------------- x [B,C,N] f32 -> xbT [B,N,C] bf16 ; seed out=x ----------------
__global__ void xpose(const float* __restrict__ x, unsigned short* __restrict__ outbT,
                      float* __restrict__ out) {
    __shared__ unsigned short tile[32][33];
    int b = blockIdx.z;
    int n0 = blockIdx.x * 32, c0 = blockIdx.y * 32;
    const float* xb = x + (size_t)b * CC * NN;
    for (int i = threadIdx.y; i < 32; i += 8) {
        int c = c0 + i, n = n0 + threadIdx.x;
        float v = xb[(size_t)c * NN + n];
        out[(size_t)b * CC * NN + (size_t)c * NN + n] = v;
        tile[i][threadIdx.x] = f2bf(v);
    }
    __syncthreads();
    for (int i = threadIdx.y; i < 32; i += 8) {
        int n = n0 + i, c = c0 + threadIdx.x;
        outbT[(size_t)b * NN * CC + (size_t)n * CC + c] = tile[threadIdx.x][i];
    }
}

// ---------------- t = relu(t_w x), p = relu(p_w x); store transposed [N,P] ----------------
// One wave per 16x16 tile; 4 waves/block cover the 4 m-tiles of one n-strip.
__global__ void feat_gemm(const unsigned short* __restrict__ xbT,
                          const unsigned short* __restrict__ twb,
                          const unsigned short* __restrict__ pwb,
                          unsigned short* __restrict__ tT, unsigned short* __restrict__ pT) {
    int b = blockIdx.y;
    int m0 = (threadIdx.x >> 5) * 16;
    int n0 = blockIdx.x * 16;
    const unsigned short* Bt = xbT + (size_t)b * NN * CC;
    v8f at = {}, ap = {};
    u16x16 b0 = load_bt_frag(Bt, CC, 0, n0);
    u16x16 t0 = load_a_frag(twb, CC, m0, 0);
    u16x16 p0 = load_a_frag(pwb, CC, m0, 0);
#pragma unroll 2
    for (int kn = 32; kn < CC; kn += 32) {
        u16x16 b1 = load_bt_frag(Bt, CC, kn, n0);
        u16x16 t1 = load_a_frag(twb, CC, m0, kn);
        u16x16 p1 = load_a_frag(pwb, CC, m0, kn);
        at = wmma_bf16(t0, b0, at);
        ap = wmma_bf16(p0, b0, ap);
        b0 = b1; t0 = t1; p0 = p1;
    }
    at = wmma_bf16(t0, b0, at);
    ap = wmma_bf16(p0, b0, ap);

    v8f ra, rp;
#pragma unroll
    for (int r = 0; r < 8; ++r) {
        ra[r] = at[r] > 0.f ? at[r] : 0.f;
        rp[r] = ap[r] > 0.f ? ap[r] : 0.f;
    }
    store_dT_bf16(tT + (size_t)b * NN * PP, PP, n0, m0, ra);
    store_dT_bf16(pT + (size_t)b * NN * PP, PP, n0, m0, rp);
}

// ---------------- att = 0.5*(t^T p + p^T t), bf16 [B,N,N] ----------------
__global__ void att_gemm(const unsigned short* __restrict__ tT,
                         const unsigned short* __restrict__ pT,
                         unsigned short* __restrict__ att) {
    int b = blockIdx.z;
    int wave = threadIdx.x >> 5;
    int n0 = blockIdx.x * 64 + (wave & 1) * 32;   // att row block
    int m0 = blockIdx.y * 64 + (wave >> 1) * 32;  // att col block
    const unsigned short* tb = tT + (size_t)b * NN * PP;
    const unsigned short* pb = pT + (size_t)b * NN * PP;
    v8f acc[2][2] = {};
#pragma unroll
    for (int k0 = 0; k0 < PP; k0 += 32) {
        u16x16 aT[2], aP[2], bT[2], bP[2];
#pragma unroll
        for (int i = 0; i < 2; ++i) {
            aT[i] = load_a_frag(tb, PP, n0 + 16 * i, k0);
            aP[i] = load_a_frag(pb, PP, n0 + 16 * i, k0);
            bP[i] = load_bt_frag(pb, PP, k0, m0 + 16 * i);
            bT[i] = load_bt_frag(tb, PP, k0, m0 + 16 * i);
        }
#pragma unroll
        for (int i = 0; i < 2; ++i)
#pragma unroll
            for (int j = 0; j < 2; ++j) {
                acc[i][j] = wmma_bf16(aT[i], bP[j], acc[i][j]);
                acc[i][j] = wmma_bf16(aP[i], bT[j], acc[i][j]);
            }
    }
    unsigned short* ab = att + (size_t)b * NN * NN;
#pragma unroll
    for (int i = 0; i < 2; ++i)
#pragma unroll
        for (int j = 0; j < 2; ++j) {
            v8f h;
#pragma unroll
            for (int r = 0; r < 8; ++r) h[r] = 0.5f * acc[i][j][r];
            // transposed store == att (symmetric)
            store_dT_bf16(ab, NN, m0 + 16 * j, n0 + 16 * i, h);
        }
}

// ---------------- deg row-sum -> dinv ----------------
__global__ void deg_kernel(const unsigned short* __restrict__ att, float* __restrict__ dinv) {
    int wave = threadIdx.x >> 5, lane = threadIdx.x & 31;
    int row = blockIdx.x * 4 + wave;  // 0 .. B*N-1
    const unsigned short* r = att + (size_t)row * NN;
    float s = 0.f;
    for (int c = lane * 8; c < NN; c += 32 * 8) {
        u16x8 v = *(const u16x8*)(r + c);
#pragma unroll
        for (int i = 0; i < 8; ++i) s += bf2f(v[i]);
    }
#pragma unroll
    for (int off = 16; off > 0; off >>= 1) s += __shfl_xor(s, off, 32);
    if (lane == 0) dinv[row] = (s != 0.f) ? rsqrtf(s) : 0.f;
}

// ---------------- att *= dinv[n]*dinv[m], in place ----------------
__global__ void norm_att(unsigned short* __restrict__ att, const float* __restrict__ dinv) {
    int row = blockIdx.y;  // b*N + n
    int b = row >> 12;
    float dn = dinv[row];
    const float* dm = dinv + (size_t)b * NN;
    unsigned short* r = att + (size_t)row * NN;
    int idx = (blockIdx.x * blockDim.x + threadIdx.x) * 8;
    if (idx < NN) {
        u16x8 v = *(const u16x8*)(r + idx);
        u16x8 o;
#pragma unroll
        for (int i = 0; i < 8; ++i) o[i] = f2bf(bf2f(v[i]) * dn * dm[idx + i]);
        *(u16x8*)(r + idx) = o;
    }
}

// ---------------- g = g_w[s] @ out ; store g [P,N] + gT [N,P] ----------------
__global__ void g_gemm(const unsigned short* __restrict__ outbT,
                       const unsigned short* __restrict__ gw,
                       unsigned short* __restrict__ g, unsigned short* __restrict__ gT) {
    int b = blockIdx.y;
    int lane = threadIdx.x & 31;
    int m0 = (threadIdx.x >> 5) * 16;
    int n0 = blockIdx.x * 16;
    const unsigned short* Bt = outbT + (size_t)b * NN * CC;
    v8f acc = {};
    u16x16 b0 = load_bt_frag(Bt, CC, 0, n0);
    u16x16 a0 = load_a_frag(gw, CC, m0, 0);
#pragma unroll 2
    for (int kn = 32; kn < CC; kn += 32) {
        u16x16 b1 = load_bt_frag(Bt, CC, kn, n0);
        u16x16 a1 = load_a_frag(gw, CC, m0, kn);
        acc = wmma_bf16(a0, b0, acc);
        b0 = b1; a0 = a1;
    }
    acc = wmma_bf16(a0, b0, acc);

    store_dT_bf16(gT + (size_t)b * NN * PP, PP, n0, m0, acc);
    unsigned short* gb = g + (size_t)b * PP * NN;
    int col = lane & 15, h = lane >> 4;
#pragma unroll
    for (int r = 0; r < 8; ++r)
        gb[(size_t)(m0 + r + 8 * h) * NN + n0 + col] = f2bf(acc[r]);
}

// ---------------- ag = g @ att (att symmetric => rows of att are B-columns) ----------------
__global__ void ag_gemm(const unsigned short* __restrict__ g,
                        const unsigned short* __restrict__ att,
                        unsigned short* __restrict__ agT) {
    int b = blockIdx.y;
    int m0 = (threadIdx.x >> 5) * 16;
    int n0 = blockIdx.x * 16;
    const unsigned short* A    = g   + (size_t)b * PP * NN;  // [64,4096]
    const unsigned short* attb = att + (size_t)b * NN * NN;  // symmetric
    v8f acc = {};
    u16x16 b0 = load_bt_frag(attb, NN, 0, n0);
    u16x16 a0 = load_a_frag(A, NN, m0, 0);
#pragma unroll 4
    for (int kn = 32; kn < NN; kn += 32) {
        u16x16 b1 = load_bt_frag(attb, NN, kn, n0);
        u16x16 a1 = load_a_frag(A, NN, m0, kn);
        acc = wmma_bf16(a0, b0, acc);
        b0 = b1; a0 = a1;
    }
    acc = wmma_bf16(a0, b0, acc);
    store_dT_bf16(agT + (size_t)b * NN * PP, PP, n0, m0, acc);
}

// ---------------- o = W1 ag + W2 g ; BN ; residual ; refresh out & outbT ----------------
__global__ void out_gemm(const unsigned short* __restrict__ agT,
                         const unsigned short* __restrict__ gT,
                         const unsigned short* __restrict__ w1b,
                         const unsigned short* __restrict__ w2b,
                         const float* __restrict__ scale, const float* __restrict__ shift,
                         float* __restrict__ out, unsigned short* __restrict__ outbT) {
    int b = blockIdx.z;
    int lane = threadIdx.x & 31;
    int m0 = blockIdx.y * 64 + (threadIdx.x >> 5) * 16;
    int n0 = blockIdx.x * 16;
    const unsigned short* agTb = agT + (size_t)b * NN * PP;
    const unsigned short* gTb  = gT  + (size_t)b * NN * PP;
    v8f acc = {};
    u16x16 b1c = load_bt_frag(agTb, PP, 0, n0);
    u16x16 b2c = load_bt_frag(gTb,  PP, 0, n0);
    u16x16 a1c = load_a_frag(w1b, PP, m0, 0);
    u16x16 a2c = load_a_frag(w2b, PP, m0, 0);
#pragma unroll
    for (int kn = 32; kn < PP; kn += 32) {
        u16x16 b1n = load_bt_frag(agTb, PP, kn, n0);
        u16x16 b2n = load_bt_frag(gTb,  PP, kn, n0);
        u16x16 a1n = load_a_frag(w1b, PP, m0, kn);
        u16x16 a2n = load_a_frag(w2b, PP, m0, kn);
        acc = wmma_bf16(a1c, b1c, acc);
        acc = wmma_bf16(a2c, b2c, acc);
        b1c = b1n; b2c = b2n; a1c = a1n; a2c = a2n;
    }
    acc = wmma_bf16(a1c, b1c, acc);
    acc = wmma_bf16(a2c, b2c, acc);

    int col = lane & 15, h = lane >> 4;
    float* outb = out + (size_t)b * CC * NN;
    unsigned short* obT = outbT + (size_t)b * NN * CC;
    u16x8 pk;
#pragma unroll
    for (int r = 0; r < 8; ++r) {
        int c = m0 + r + 8 * h;
        size_t oi = (size_t)c * NN + n0 + col;
        float v = acc[r] * scale[c] + shift[c] + outb[oi];
        outb[oi] = v;
        pk[r] = f2bf(v);
    }
    *(u16x8*)(obT + (size_t)(n0 + col) * CC + m0 + 8 * h) = pk;
}

extern "C" void kernel_launch(void* const* d_in, const int* in_sizes, int n_in,
                              void* d_out, int out_size, void* d_ws, size_t ws_size,
                              hipStream_t stream) {
    const float* x     = (const float*)d_in[0];
    const float* t_w   = (const float*)d_in[1];
    const float* p_w   = (const float*)d_in[2];
    const float* g_w   = (const float*)d_in[3];
    const float* w1    = (const float*)d_in[4];
    const float* w2    = (const float*)d_in[5];
    const float* gamma = (const float*)d_in[6];
    const float* beta  = (const float*)d_in[7];
    const float* rm    = (const float*)d_in[8];
    const float* rv    = (const float*)d_in[9];
    float* out = (float*)d_out;  // [B,C,N] running residual, final result

    char* w = (char*)d_ws;
    auto carve = [&](size_t bytes) -> void* {
        void* p = (void*)w;
        w += (bytes + 255) & ~(size_t)255;
        return p;
    };
    unsigned short* outbT = (unsigned short*)carve((size_t)BB * NN * CC * 2);  // 8 MB
    unsigned short* tT    = (unsigned short*)carve((size_t)BB * NN * PP * 2);  // 2 MB
    unsigned short* pT    = (unsigned short*)carve((size_t)BB * NN * PP * 2);
    unsigned short* gbuf  = (unsigned short*)carve((size_t)BB * PP * NN * 2);
    unsigned short* gT    = (unsigned short*)carve((size_t)BB * NN * PP * 2);
    unsigned short* agT   = (unsigned short*)carve((size_t)BB * NN * PP * 2);
    unsigned short* att   = (unsigned short*)carve((size_t)BB * NN * NN * 2);  // 128 MB
    float* dinv  = (float*)carve((size_t)BB * NN * 4);
    unsigned short* twb = (unsigned short*)carve((size_t)PP * CC * 2);
    unsigned short* pwb = (unsigned short*)carve((size_t)PP * CC * 2);
    unsigned short* gwb = (unsigned short*)carve((size_t)SS * PP * CC * 2);
    unsigned short* w1b = (unsigned short*)carve((size_t)SS * CC * PP * 2);
    unsigned short* w2b = (unsigned short*)carve((size_t)SS * CC * PP * 2);
    float* scale = (float*)carve((size_t)SS * CC * 4);
    float* shift = (float*)carve((size_t)SS * CC * 4);

    prep_params<<<64, 256, 0, stream>>>(t_w, p_w, g_w, w1, w2, gamma, beta, rm, rv,
                                        twb, pwb, gwb, w1b, w2b, scale, shift);
    xpose<<<dim3(NN / 32, CC / 32, BB), dim3(32, 8), 0, stream>>>(x, outbT, out);
    feat_gemm<<<dim3(NN / 16, BB), 128, 0, stream>>>(outbT, twb, pwb, tT, pT);
    att_gemm<<<dim3(NN / 64, NN / 64, BB), 128, 0, stream>>>(tT, pT, att);
    deg_kernel<<<BB * NN / 4, 128, 0, stream>>>(att, dinv);
    norm_att<<<dim3(NN / (8 * 128), BB * NN), 128, 0, stream>>>(att, dinv);

    for (int s = 0; s < SS; ++s) {
        g_gemm<<<dim3(NN / 16, BB), 128, 0, stream>>>(outbT, gwb + (size_t)s * PP * CC,
                                                      gbuf, gT);
        ag_gemm<<<dim3(NN / 16, BB), 128, 0, stream>>>(gbuf, att, agT);
        out_gemm<<<dim3(NN / 16, CC / 64, BB), 128, 0, stream>>>(
            agT, gT, w1b + (size_t)s * CC * PP, w2b + (size_t)s * CC * PP,
            scale + (size_t)s * CC, shift + (size_t)s * CC, out, outbT);
    }
    (void)in_sizes; (void)n_in; (void)out_size; (void)ws_size;
}
